// MultiHeadAttention_72361609003594
// MI455X (gfx1250) — matrix operand order repaired and checked by reading, not compiled, not run on hardware
//
#include <hip/hip_runtime.h>
#include <hip/hip_bf16.h>

// Problem constants (match reference)
#define NB  2
#define NS  512
#define ND  512
#define NH  8
#define NHD 64
#define BS  (NB * NS)   // 1024

typedef __attribute__((ext_vector_type(2)))  float  v2f;
typedef __attribute__((ext_vector_type(8)))  float  v8f;
typedef __attribute__((ext_vector_type(16))) __bf16 v16bf;

// ---------------------------------------------------------------------------
// Tensor Data Mover (CDNA5): DMA a 2D bf16 tile Global -> LDS.
// D# encoding per ISA cdna5 ch.8: group0 = {flags/count, lds_addr, global_addr,
// type}, group1 = {data_size/pad, tensor dims, tile dims, dim0 stride}.
// Each 64B tile row is padded with 16B in LDS (pad_interval=16DW, pad_amount=
// 4DW) -> LDS row stride 80B = 40 bf16, which both de-conflicts LDS banks and
// keeps 16B alignment for the WMMA fragment reads.
// ---------------------------------------------------------------------------
#if __has_builtin(__builtin_amdgcn_tensor_load_to_lds) && \
    __has_builtin(__builtin_amdgcn_s_wait_tensorcnt)
#define HAVE_TDM 1
typedef __attribute__((ext_vector_type(4))) unsigned int tdm_u32x4;
typedef __attribute__((ext_vector_type(8))) int          tdm_i32x8;
typedef __attribute__((ext_vector_type(4))) int          tdm_i32x4;

__device__ __forceinline__ void tdm_load_tile_bf16(
    const void* gsrc, unsigned lds_off, unsigned tile_w_elems,
    unsigned tile_h, unsigned row_stride_elems)
{
  unsigned long long ga = (unsigned long long)(uintptr_t)gsrc;
  tdm_u32x4 g0;
  g0[0] = 1u;                                  // count=1, user descriptor
  g0[1] = lds_off;                             // D#.lds_addr (bytes)
  g0[2] = (unsigned)ga;                        // global_addr[31:0]
  g0[3] = (unsigned)((ga >> 32) & 0x01ffffffu) // global_addr[56:32]
        | (2u << 30);                          // type = 2 ("image")
  const unsigned TD = 0x100000u;               // generous tensor dims (no clip)
  tdm_i32x8 g1;
  g1[0] = (int)((1u << 16)                     // data_size = 1 -> 2 bytes
              | (1u << 20)                     // pad_enable
              | (3u << 22)                     // pad_interval: 16 DWORDs (64B)
              | (3u << 25));                   // pad_amount:    4 DWORDs (16B)
  g1[1] = (int)((TD & 0xffffu) << 16);                          // dim0[15:0]
  g1[2] = (int)(((TD >> 16) & 0xffffu) | ((TD & 0xffffu) << 16)); // dim0hi|dim1lo
  g1[3] = (int)(((TD >> 16) & 0xffffu) |
                ((tile_w_elems & 0xffffu) << 16));              // dim1hi|tile0
  g1[4] = (int)(tile_h & 0xffffu);             // tile_dim1 (tile_dim2 = 0)
  g1[5] = (int)row_stride_elems;               // tensor_dim0_stride[31:0]
  g1[6] = 0;
  g1[7] = 0;
  tdm_i32x4 z4 = {};
#if defined(__clang_major__) && (__clang_major__ >= 23)
  tdm_i32x8 z8 = {};
  __builtin_amdgcn_tensor_load_to_lds(g0, g1, z4, z4, z8, 0);
#else
  __builtin_amdgcn_tensor_load_to_lds(g0, g1, z4, z4, 0);
#endif
}
#else
#define HAVE_TDM 0
#endif

__device__ __forceinline__ float fast_tanh(float x) {
  // tanh(x) = (e^{2x}-1)/(e^{2x}+1); clamp keeps e finite (one v_exp_f32)
  x = fminf(fmaxf(x, -15.0f), 15.0f);
  float e = __expf(2.0f * x);
  return (e - 1.0f) / (e + 1.0f);
}

// ---------------------------------------------------------------------------
// K0: fold per-head matrices into projection weights.
//   Wdst[i, h*64+d] = sum_e Wsrc[i, h*64+e] * Amat[e,d];  row ND folds bias.
// ---------------------------------------------------------------------------
__global__ __launch_bounds__(256) void fold_weights(
    const float* __restrict__ Wsrc, const float* __restrict__ Amat,
    const float* __restrict__ bsrc, float* __restrict__ Wdst,
    float* __restrict__ bdst)
{
  __shared__ float Ash[NHD][NHD];   // 16 KB
  for (int i = threadIdx.x; i < NHD * NHD; i += 256)
    Ash[i / NHD][i % NHD] = Amat[i];
  __syncthreads();

  int i = blockIdx.x;               // 0..ND-1 = weight row, ND = bias
  for (int c = threadIdx.x; c < ND; c += 256) {
    int h = c / NHD, d = c % NHD;
    const float* src = (i == ND) ? (bsrc + h * NHD)
                                 : (Wsrc + (size_t)i * ND + h * NHD);
    float acc = 0.0f;
    #pragma unroll 8
    for (int e = 0; e < NHD; ++e) acc += src[e] * Ash[e][d];
    if (i == ND) bdst[c] = acc;
    else         Wdst[(size_t)i * ND + c] = acc;
  }
}

// ---------------------------------------------------------------------------
// K1: fp32 GEMM + bias via V_WMMA_F32_16X16X4_F32 (precision-critical path).
//   Block = 128 threads = 4 waves, 64x64 tile; wave w -> rows [16w,16w+16).
// ---------------------------------------------------------------------------
__global__ __launch_bounds__(128) void gemm_f32_wmma(
    const float* __restrict__ A, const float* __restrict__ W,
    const float* __restrict__ bias, float* __restrict__ C,
    int M, int N, int K)
{
  __shared__ float As[64][17];     // 64 x 16 tile (+pad)
  __shared__ float Ws[16][65];     // 16 x 64 tile (+pad)

  const int tid  = threadIdx.x;
  const int lane = tid & 31, wave = tid >> 5;
  const int lh   = lane >> 4, l16 = lane & 15;
  const int rowBlk = blockIdx.y * 64;
  const int colBlk = blockIdx.x * 64;

  v8f acc[4] = {};

  for (int k0 = 0; k0 < K; k0 += 16) {
    for (int i = tid; i < 64 * 16; i += 128) {
      int r = i >> 4, c = i & 15;
      As[r][c] = A[(size_t)(rowBlk + r) * K + k0 + c];
    }
    for (int i = tid; i < 16 * 64; i += 128) {
      int r = i >> 6, c = i & 63;
      Ws[r][c] = W[(size_t)(k0 + r) * N + colBlk + c];
    }
    if (k0 + 16 < K)   // prefetch next A tile row for this thread
      __builtin_prefetch(A + (size_t)(rowBlk + (tid >> 4)) * K + k0 + 16, 0, 1);
    __syncthreads();

    #pragma unroll
    for (int kk = 0; kk < 16; kk += 4) {
      // 32-bit A 16x4 layout: lanes 0-15 K={0,1}, lanes 16-31 K={2,3}
      int ka = kk + lh * 2;
      v2f a;
      a.x = As[wave * 16 + l16][ka];
      a.y = As[wave * 16 + l16][ka + 1];
      #pragma unroll
      for (int t = 0; t < 4; ++t) {
        v2f b;
        b.x = Ws[ka][t * 16 + l16];
        b.y = Ws[ka + 1][t * 16 + l16];
        acc[t] = __builtin_amdgcn_wmma_f32_16x16x4_f32(
            false, a, false, b, (short)0, acc[t], false, false);
      }
    }
    __syncthreads();
  }

  #pragma unroll
  for (int t = 0; t < 4; ++t) {
    int col = colBlk + t * 16 + l16;
    float bv = bias ? bias[col] : 0.0f;
    #pragma unroll
    for (int r = 0; r < 8; ++r) {
      int row = rowBlk + wave * 16 + lh * 8 + r;   // C/D: M = r + 8*half
      C[(size_t)row * N + col] = acc[t][r] + bv;
    }
  }
}

// ---------------------------------------------------------------------------
// K2: bf16 GEMM + bias via V_WMMA_F32_16X16X32_BF16 (fp32 in/out, bf16 MAC).
//     B tile kept transposed in LDS so fragment reads are two contiguous 16B runs.
// ---------------------------------------------------------------------------
__global__ __launch_bounds__(128) void gemm_bias_bf16wmma(
    const float* __restrict__ A, const float* __restrict__ W,
    const float* __restrict__ bias, float* __restrict__ C,
    int M, int N, int K)
{
  __shared__ __bf16 As [64][40];   // [m][k], 80B row stride
  __shared__ __bf16 WsT[64][40];   // [n][k] transposed

  const int tid  = threadIdx.x;
  const int lane = tid & 31, wave = tid >> 5;
  const int lh   = lane >> 4, l16 = lane & 15;
  const int rowBlk = blockIdx.y * 64;
  const int colBlk = blockIdx.x * 64;

  v8f acc[4] = {};

  for (int k0 = 0; k0 < K; k0 += 32) {
    for (int i = tid; i < 64 * 32; i += 128) {
      int r = i >> 5, c = i & 31;
      As[r][c] = (__bf16)A[(size_t)(rowBlk + r) * K + k0 + c];
    }
    for (int i = tid; i < 32 * 64; i += 128) {
      int r = i >> 6, c = i & 63;                 // r = k, c = n (coalesced)
      WsT[c][r] = (__bf16)W[(size_t)(k0 + r) * N + colBlk + c];
    }
    __syncthreads();

    // 16-bit A 16x32 fragment: K(e) = half*8 + (e<8 ? e : e+8)
    v16bf a;
    #pragma unroll
    for (int e = 0; e < 16; ++e) {
      int ka = lh * 8 + (e < 8 ? e : e + 8);
      a[e] = As[wave * 16 + l16][ka];
    }
    #pragma unroll
    for (int t = 0; t < 4; ++t) {
      v16bf b;   // B fragment: N = lane&15, K(e) = half*8 + (e<8 ? e : e+8)
      #pragma unroll
      for (int e = 0; e < 16; ++e)
        b[e] = WsT[t * 16 + l16][lh * 8 + (e < 8 ? e : e + 8)];
      acc[t] = __builtin_amdgcn_wmma_f32_16x16x32_bf16(
          false, a, false, b, (short)0, acc[t], false, false);
    }
    __syncthreads();
  }

  #pragma unroll
  for (int t = 0; t < 4; ++t) {
    int col = colBlk + t * 16 + l16;
    float bv = bias ? bias[col] : 0.0f;
    #pragma unroll
    for (int r = 0; r < 8; ++r) {
      int row = rowBlk + wave * 16 + lh * 8 + r;
      C[(size_t)row * N + col] = acc[t][r] + bv;
    }
  }
}

// ---------------------------------------------------------------------------
// K3: repack projected V into TRANSPOSED per-head bf16: vbT[b,h,d,k].
//     (d-major so the attn@V B-tile is the same 2D TDM shape as the A-tile.)
// ---------------------------------------------------------------------------
__global__ void v_to_bf16_t(const float* __restrict__ vproj,
                            __bf16* __restrict__ vbT)
{
  int idx = blockIdx.x * blockDim.x + threadIdx.x;    // over B*H*HD*S
  if (idx >= NB * NS * ND) return;
  int k  = idx & (NS - 1);
  int d  = (idx >> 9) & (NHD - 1);
  int bh = idx >> 15;
  int b = bh >> 3, h = bh & 7;
  vbT[idx] = (__bf16)vproj[((size_t)(b * NS + k)) * ND + h * NHD + d];
}

// ---------------------------------------------------------------------------
// K4: Bahdanau scores + softmax.  One 256-thread block per (b,h,q) row.
//     Dominant cost of the whole op: 64 tanh per k column, 512 columns.
// ---------------------------------------------------------------------------
__global__ __launch_bounds__(256) void scores_softmax(
    const float* __restrict__ qt, const float* __restrict__ kt,
    const float* __restrict__ av, float* __restrict__ attn_out,
    __bf16* __restrict__ attn_bf)
{
  const int bhq = blockIdx.x;          // ((b*H + h)*S + q)
  const int q   = bhq & (NS - 1);
  const int bh  = bhq >> 9;
  const int h   = bh & (NH - 1), b = bh >> 3;
  const int tid = threadIdx.x;

  __shared__ float qrow[NHD], avs[NHD];
  __shared__ float red[256];

  if (tid < NHD) {
    qrow[tid] = qt[((size_t)(b * NS + q)) * ND + h * NHD + tid];
    avs[tid]  = av[tid];
  }
  __syncthreads();

  float sc[2];
  #pragma unroll
  for (int j = 0; j < 2; ++j) {
    const int k = tid + j * 256;
    const float4* kr4 = reinterpret_cast<const float4*>(
        kt + ((size_t)(b * NS + k)) * ND + h * NHD);
    float s = 0.0f;
    #pragma unroll
    for (int d4 = 0; d4 < NHD / 4; ++d4) {
      float4 kv = kr4[d4];
      s += fast_tanh(qrow[d4 * 4 + 0] + kv.x) * avs[d4 * 4 + 0];
      s += fast_tanh(qrow[d4 * 4 + 1] + kv.y) * avs[d4 * 4 + 1];
      s += fast_tanh(qrow[d4 * 4 + 2] + kv.z) * avs[d4 * 4 + 2];
      s += fast_tanh(qrow[d4 * 4 + 3] + kv.w) * avs[d4 * 4 + 3];
    }
    sc[j] = s;
  }

  // row max
  red[tid] = fmaxf(sc[0], sc[1]);
  __syncthreads();
  for (int off = 128; off > 0; off >>= 1) {
    if (tid < off) red[tid] = fmaxf(red[tid], red[tid + off]);
    __syncthreads();
  }
  const float rowmax = red[0];
  __syncthreads();

  // exp + sum
  const float e0 = __expf(sc[0] - rowmax);
  const float e1 = __expf(sc[1] - rowmax);
  red[tid] = e0 + e1;
  __syncthreads();
  for (int off = 128; off > 0; off >>= 1) {
    if (tid < off) red[tid] += red[tid + off];
    __syncthreads();
  }
  const float inv = 1.0f / red[0];

  const size_t base = (size_t)bhq * NS;
  const float w0 = e0 * inv, w1 = e1 * inv;
  attn_out[base + tid]       = w0;
  attn_out[base + tid + 256] = w1;
  attn_bf[base + tid]        = (__bf16)w0;
  attn_bf[base + tid + 256]  = (__bf16)w1;
}

// ---------------------------------------------------------------------------
// K5: ctx = attn @ V per (b,h) head, bf16 WMMA with TDM tile staging.
//     A = attn_bf [bh][S][S], B = vbT [bh][HD][S]; both tiles are 64x32 bf16
//     with row stride S -> one tensor_load_to_lds each per K step.
// ---------------------------------------------------------------------------
__global__ __launch_bounds__(128) void ctx_gemm_bf16(
    const __bf16* __restrict__ attnb, const __bf16* __restrict__ vbT,
    float* __restrict__ ctx)
{
  __shared__ __bf16 As[64][40];   // [q][k] tile, 64B rows + 16B TDM pad
  __shared__ __bf16 Bt[64][40];   // [d][k] tile (B transposed)

  const int bh = blockIdx.x;           // 0..15
  const int b  = bh >> 3, h = bh & 7;
  const int rowBlk = blockIdx.y * 64;  // q tile

  const __bf16* Abase = attnb + (size_t)bh * NS * NS + (size_t)rowBlk * NS;
  const __bf16* Bbase = vbT   + (size_t)bh * NHD * NS;

  const int tid  = threadIdx.x;
  const int lane = tid & 31, wave = tid >> 5;
  const int lh   = lane >> 4, l16 = lane & 15;

  v8f acc[4] = {};

  for (int k0 = 0; k0 < NS; k0 += 32) {
#if HAVE_TDM
    if (wave == 0) {   // TDM is per-wave (EXEC ignored): one wave DMAs both tiles
      tdm_load_tile_bf16(Abase + k0, (unsigned)(uintptr_t)&As[0][0], 32, 64, NS);
      tdm_load_tile_bf16(Bbase + k0, (unsigned)(uintptr_t)&Bt[0][0], 32, 64, NS);
      __builtin_amdgcn_s_wait_tensorcnt(0);
    }
#else
    for (int i = tid; i < 64 * 32; i += 128) {
      int r = i >> 5, c = i & 31;
      As[r][c] = Abase[(size_t)r * NS + k0 + c];
      Bt[r][c] = Bbase[(size_t)r * NS + k0 + c];
    }
#endif
    __syncthreads();

    v16bf a;
    #pragma unroll
    for (int e = 0; e < 16; ++e) {
      int ka = lh * 8 + (e < 8 ? e : e + 8);
      a[e] = As[wave * 16 + l16][ka];
    }
    #pragma unroll
    for (int t = 0; t < 4; ++t) {
      v16bf bfrag;
      #pragma unroll
      for (int e = 0; e < 16; ++e)
        bfrag[e] = Bt[t * 16 + l16][lh * 8 + (e < 8 ? e : e + 8)];
      acc[t] = __builtin_amdgcn_wmma_f32_16x16x32_bf16(
          false, a, false, bfrag, (short)0, acc[t], false, false);
    }
    __syncthreads();
  }

  #pragma unroll
  for (int t = 0; t < 4; ++t) {
    int col = t * 16 + l16;            // d within head
    #pragma unroll
    for (int r = 0; r < 8; ++r) {
      int row = rowBlk + wave * 16 + lh * 8 + r;   // q index
      ctx[((size_t)(b * NS + row)) * ND + h * NHD + col] = acc[t][r];
    }
  }
}

// ---------------------------------------------------------------------------
// Host launcher
// ---------------------------------------------------------------------------
extern "C" void kernel_launch(void* const* d_in, const int* in_sizes, int n_in,
                              void* d_out, int out_size, void* d_ws, size_t ws_size,
                              hipStream_t stream) {
  (void)in_sizes; (void)n_in; (void)out_size; (void)ws_size;

  const float* query = (const float*)d_in[0];
  const float* key_  = (const float*)d_in[1];
  const float* value = (const float*)d_in[2];
  const float* Wq    = (const float*)d_in[3];
  const float* bq    = (const float*)d_in[4];
  const float* Wk    = (const float*)d_in[5];
  const float* bk    = (const float*)d_in[6];
  const float* Wv    = (const float*)d_in[7];
  const float* bv    = (const float*)d_in[8];
  const float* Wo    = (const float*)d_in[9];
  const float* bo    = (const float*)d_in[10];
  const float* Aq    = (const float*)d_in[11];
  const float* Ak    = (const float*)d_in[12];
  const float* av    = (const float*)d_in[13];

  float* ws = (float*)d_ws;
  float*  qt      = ws;                       // [1024,512]
  float*  ktm     = ws + 524288;              // [1024,512]
  float*  vproj   = ws + 1048576;             // [1024,512]
  float*  ctx     = ws + 1572864;             // [1024,512]
  float*  Wq2     = ws + 2097152;             // [512,512]
  float*  Wk2     = ws + 2359296;             // [512,512]
  float*  bq2     = ws + 2621440;             // [512]
  float*  bk2     = ws + 2621952;             // [512]
  __bf16* vbT     = (__bf16*)(ws + 2622464);  // [16,64,512] bf16 (d-major)
  __bf16* attnbf  = (__bf16*)(ws + 2884608);  // [16,512,512] bf16

  float* out  = (float*)d_out;        // [2,512,512]
  float* attn = out + NB * NS * ND;   // [2,8,512,512]

  // 1) fold Aq/Ak into the projection weights (q/k never materialized)
  fold_weights<<<ND + 1, 256, 0, stream>>>(Wq, Aq, bq, Wq2, bq2);
  fold_weights<<<ND + 1, 256, 0, stream>>>(Wk, Ak, bk, Wk2, bk2);

  // 2) fp32-WMMA projections (precision-critical, pre-softmax)
  dim3 gP(ND / 64, BS / 64);          // (8, 16)
  gemm_f32_wmma<<<gP, 128, 0, stream>>>(query, Wq2, bq2, qt,    BS, ND, ND);
  gemm_f32_wmma<<<gP, 128, 0, stream>>>(key_,  Wk2, bk2, ktm,   BS, ND, ND);
  gemm_f32_wmma<<<gP, 128, 0, stream>>>(value, Wv,  bv,  vproj, BS, ND, ND);

  // 3) repack V as transposed per-head bf16
  v_to_bf16_t<<<(NB * NS * ND) / 256, 256, 0, stream>>>(vproj, vbT);

  // 4) tanh scores + softmax (dominant stage, trans-op bound)
  scores_softmax<<<NB * NH * NS, 256, 0, stream>>>(qt, ktm, av, attn, attnbf);

  // 5) ctx = attn @ V per head (bf16 WMMA, TDM-staged tiles)
  ctx_gemm_bf16<<<dim3(NB * NH, NS / 64), 128, 0, stream>>>(attnbf, vbT, ctx);

  // 6) output projection (bf16 WMMA)
  gemm_bias_bf16wmma<<<gP, 128, 0, stream>>>(ctx, Wo, bo, out, BS, ND, ND);
}
